// Embedding_24206435680939
// MI455X (gfx1250) — matrix operand order repaired and checked by reading, not compiled, not run on hardware
//
#include <hip/hip_runtime.h>
#include <hip/hip_bf16.h>
#include <math.h>

// ---------------- constants ----------------
#define V_KP      6890
#define N_PTS     16384          // 256 rays * 64 points
#define D_IN      700
#define D_PAD     704            // 22 * 32
#define KT1       22             // 704/32
#define KT23      8              // 256/32
#define D_HID     256
#define D_OUT_ROW 283            // 256 + 27
#define CHUNK     2048

typedef __attribute__((ext_vector_type(16))) __bf16 v16bf;
typedef __attribute__((ext_vector_type(8)))  float  v8f;
typedef __attribute__((ext_vector_type(4)))  int    v4i;

#define AS1 __attribute__((address_space(1)))
#define AS3 __attribute__((address_space(3)))

typedef AS1 v4i* gv4i_p;   // global int4 pointer (async src)
typedef AS3 v4i* sv4i_p;   // LDS int4 pointer   (async dst)

#if __has_builtin(__builtin_amdgcn_global_load_async_to_lds_b128)
#define USE_ASYNC_LDS 1
#else
#define USE_ASYNC_LDS 0
#endif

union Frag {
    v16bf    v;
    unsigned d[8];
    uint4    q[2];
};

__device__ __forceinline__ unsigned short f2bf(float x) {
    unsigned u = __float_as_uint(x);
    unsigned r = u + 0x7fffu + ((u >> 16) & 1u);   // round-to-nearest-even
    return (unsigned short)(r >> 16);
}

__device__ __forceinline__ void wait_async0() {
#if __has_builtin(__builtin_amdgcn_s_wait_asynccnt)
    __builtin_amdgcn_s_wait_asynccnt(0);
#else
    asm volatile("s_wait_asynccnt 0x0" ::: "memory");
#endif
}

// ---------------- kernel 0: per-vertex 4x4 inverse, keep top-left 3x3 ----------------
__global__ void invert_kernel(const float* __restrict__ trans_kp,
                              float* __restrict__ Rws, int V) {
    int n = blockIdx.x * 256 + threadIdx.x;
    if (n >= V) return;
    float m[16];
#pragma unroll
    for (int i = 0; i < 16; ++i) m[i] = trans_kp[(size_t)i * V + n];

    float inv[16];
    inv[0]  =  m[5]*m[10]*m[15] - m[5]*m[11]*m[14] - m[9]*m[6]*m[15] + m[9]*m[7]*m[14] + m[13]*m[6]*m[11] - m[13]*m[7]*m[10];
    inv[4]  = -m[4]*m[10]*m[15] + m[4]*m[11]*m[14] + m[8]*m[6]*m[15] - m[8]*m[7]*m[14] - m[12]*m[6]*m[11] + m[12]*m[7]*m[10];
    inv[8]  =  m[4]*m[9]*m[15]  - m[4]*m[11]*m[13] - m[8]*m[5]*m[15] + m[8]*m[7]*m[13] + m[12]*m[5]*m[11] - m[12]*m[7]*m[9];
    inv[12] = -m[4]*m[9]*m[14]  + m[4]*m[10]*m[13] + m[8]*m[5]*m[14] - m[8]*m[6]*m[13] - m[12]*m[5]*m[10] + m[12]*m[6]*m[9];
    inv[1]  = -m[1]*m[10]*m[15] + m[1]*m[11]*m[14] + m[9]*m[2]*m[15] - m[9]*m[3]*m[14] - m[13]*m[2]*m[11] + m[13]*m[3]*m[10];
    inv[5]  =  m[0]*m[10]*m[15] - m[0]*m[11]*m[14] - m[8]*m[2]*m[15] + m[8]*m[3]*m[14] + m[12]*m[2]*m[11] - m[12]*m[3]*m[10];
    inv[9]  = -m[0]*m[9]*m[15]  + m[0]*m[11]*m[13] + m[8]*m[1]*m[15] - m[8]*m[3]*m[13] - m[12]*m[1]*m[11] + m[12]*m[3]*m[9];
    inv[13] =  m[0]*m[9]*m[14]  - m[0]*m[10]*m[13] - m[8]*m[1]*m[14] + m[8]*m[2]*m[13] + m[12]*m[1]*m[10] - m[12]*m[2]*m[9];
    inv[2]  =  m[1]*m[6]*m[15]  - m[1]*m[7]*m[14]  - m[5]*m[2]*m[15] + m[5]*m[3]*m[14] + m[13]*m[2]*m[7]  - m[13]*m[3]*m[6];
    inv[6]  = -m[0]*m[6]*m[15]  + m[0]*m[7]*m[14]  + m[4]*m[2]*m[15] - m[4]*m[3]*m[14] - m[12]*m[2]*m[7]  + m[12]*m[3]*m[6];
    inv[10] =  m[0]*m[5]*m[15]  - m[0]*m[7]*m[13]  - m[4]*m[1]*m[15] + m[4]*m[3]*m[13] + m[12]*m[1]*m[7]  - m[12]*m[3]*m[5];
    inv[14] = -m[0]*m[5]*m[14]  + m[0]*m[6]*m[13]  + m[4]*m[1]*m[14] - m[4]*m[2]*m[13] - m[12]*m[1]*m[6]  + m[12]*m[2]*m[5];

    float det = m[0]*inv[0] + m[1]*inv[4] + m[2]*inv[8] + m[3]*inv[12];
    float rdet = 1.0f / det;
#pragma unroll
    for (int i = 0; i < 3; ++i)
#pragma unroll
        for (int j = 0; j < 3; ++j)
            Rws[(size_t)n * 9 + i * 3 + j] = inv[i * 4 + j] * rdet;
}

// ---------------- kernel 1: repack f32 weights -> bf16 B-fragment layout ----------------
// B fragment (32x16 bf16, contiguous K order, same as A reads): lane l holds column N=(l&15);
//   lanes 0-15 -> K=0..15, lanes 16-31 -> K=16..31; VGPR v -> K = base + 2v, 2v+1.
// Storage: Wpk[((kt*16 + nt)*32 + lane)*16 + 2v + {0,1}]
__global__ void repack_kernel(const float* __restrict__ W, int K, int kTiles,
                              unsigned short* __restrict__ Wpk) {
    int t = blockIdx.x * 256 + threadIdx.x;
    int total = kTiles * 16 * 32;
    if (t >= total) return;
    int lane = t & 31;
    int frag = t >> 5;
    int nt = frag & 15;
    int kt = frag >> 4;
    unsigned short* dst = Wpk + (size_t)t * 16;
    int n = nt * 16 + (lane & 15);
#pragma unroll
    for (int v = 0; v < 8; ++v) {
        int k = kt * 32 + ((lane >> 4) * 16) + v * 2;
        float a = (k     < K) ? W[(size_t)k       * 256 + n] : 0.0f;
        float b = (k + 1 < K) ? W[(size_t)(k + 1) * 256 + n] : 0.0f;
        dst[v * 2 + 0] = f2bf(a);
        dst[v * 2 + 1] = f2bf(b);
    }
}

// ---------------- kernel 2: KNN + features + direction posenc ----------------
__global__ __launch_bounds__(256)
void feature_kernel(const float* __restrict__ pts,
                    const float* __restrict__ keypoints,
                    const int*   __restrict__ neighbors,
                    const float* __restrict__ rest_pose,
                    const float* __restrict__ latent,
                    const float* __restrict__ Rws,
                    unsigned short* __restrict__ Hin,
                    float* __restrict__ out) {
    __shared__ float skp[CHUNK * 4];   // x,y,z,|k|^2
    const int n = blockIdx.x * 256 + threadIdx.x;

    const float px = pts[(size_t)n * 6 + 0];
    const float py = pts[(size_t)n * 6 + 1];
    const float pz = pts[(size_t)n * 6 + 2];
    const float pp = px * px + py * py + pz * pz;

    float best = 3.4e38f;
    int bidx = 0;
    for (int base = 0; base < V_KP; base += CHUNK) {
        int cnt = min(CHUNK, V_KP - base);
        __syncthreads();
        for (int j = threadIdx.x; j < cnt; j += 256) {
            float kx = keypoints[(size_t)(base + j) * 3 + 0];
            float ky = keypoints[(size_t)(base + j) * 3 + 1];
            float kz = keypoints[(size_t)(base + j) * 3 + 2];
            skp[j * 4 + 0] = kx; skp[j * 4 + 1] = ky; skp[j * 4 + 2] = kz;
            skp[j * 4 + 3] = kx * kx + ky * ky + kz * kz;
        }
        __syncthreads();
        for (int j = 0; j < cnt; ++j) {
            float d2 = pp - 2.0f * (px * skp[j * 4] + py * skp[j * 4 + 1] + pz * skp[j * 4 + 2])
                       + skp[j * 4 + 3];
            if (d2 < best) { best = d2; bidx = base + j; }   // first-min like argmin
        }
    }

    // direction = R[knn] @ (p - kp[knn]), normalized
    float dx = px - keypoints[(size_t)bidx * 3 + 0];
    float dy = py - keypoints[(size_t)bidx * 3 + 1];
    float dz = pz - keypoints[(size_t)bidx * 3 + 2];
    float dir[3];
#pragma unroll
    for (int i = 0; i < 3; ++i) {
        const float* R = Rws + (size_t)bidx * 9 + i * 3;
        dir[i] = R[0] * dx + R[1] * dy + R[2] * dz;
    }
    float nrm = sqrtf(dir[0] * dir[0] + dir[1] * dir[1] + dir[2] * dir[2]);
    float rn = 1.0f / fmaxf(nrm, 1e-12f);
    dir[0] *= rn; dir[1] *= rn; dir[2] *= rn;

    // direction posenc (res=4) -> output columns 256..282 (f32)
    {
        float* orow = out + (size_t)n * D_OUT_ROW;
#pragma unroll
        for (int i = 0; i < 3; ++i) orow[256 + i] = dir[i];
#pragma unroll
        for (int f = 0; f < 4; ++f) {
            float fr = (float)(1 << f);
#pragma unroll
            for (int i = 0; i < 3; ++i) {
                orow[259 + f * 3 + i] = sinf(dir[i] * fr);
                orow[271 + f * 3 + i] = cosf(dir[i] * fr);
            }
        }
    }

    // vfeat (28) = rest_pose[vidx] (21) ++ ||p - kp[vidx]|| (7)
    int vidx[7];
    float vf[28];
#pragma unroll
    for (int j = 0; j < 7; ++j) {
        int vj = neighbors[(size_t)bidx * 7 + j];
        vidx[j] = vj;
        float kx = keypoints[(size_t)vj * 3 + 0];
        float ky = keypoints[(size_t)vj * 3 + 1];
        float kz = keypoints[(size_t)vj * 3 + 2];
        vf[3 * j + 0] = rest_pose[(size_t)vj * 3 + 0];
        vf[3 * j + 1] = rest_pose[(size_t)vj * 3 + 1];
        vf[3 * j + 2] = rest_pose[(size_t)vj * 3 + 2];
        float ex = px - kx, ey = py - ky, ez = pz - kz;
        vf[21 + j] = sqrtf(ex * ex + ey * ey + ez * ez);
    }

    // MLP input row (bf16, padded to 704): [vf(28) | sin f-major(280) | cos(280) | latent(112) | 0(4)]
    unsigned short* hrow = Hin + (size_t)n * D_PAD;
#pragma unroll
    for (int c = 0; c < 28; ++c) hrow[c] = f2bf(vf[c]);
    for (int f = 0; f < 10; ++f) {
        float fr = (float)(1 << f);
#pragma unroll
        for (int c = 0; c < 28; ++c) {
            float xf = vf[c] * fr;
            hrow[28  + f * 28 + c] = f2bf(__sinf(xf));
            hrow[308 + f * 28 + c] = f2bf(__cosf(xf));
        }
    }
#pragma unroll
    for (int j = 0; j < 7; ++j) {
        const float* lat = latent + (size_t)vidx[j] * 16;
#pragma unroll
        for (int t = 0; t < 16; ++t) hrow[588 + j * 16 + t] = f2bf(lat[t]);
    }
#pragma unroll
    for (int c = 700; c < 704; ++c) hrow[c] = 0;
}

// ---------------- kernel 3: bf16 WMMA GEMM  C = act(A @ W + b) ----------------
// block = 256 thr (8 waves), block tile 128M x 128N, wave tile 32M x 64N (8 accum frags)
// A tile double-buffered in LDS (2 x 128 rows x 32 bf16, padded stride 40 shorts = 80 B,
// conflict-free fragment reads, 16B-aligned b128 stores). Staging uses the gfx1250
// async memory->LDS path (ASYNCcnt) when available.
__global__ __launch_bounds__(256)
void gemm_wmma(const unsigned short* __restrict__ A, int lda, int kTiles,
               const unsigned short* __restrict__ Wpk,
               const float* __restrict__ bias,
               unsigned short* __restrict__ outBf,
               float* __restrict__ outF, int outStride, int mode) {
    __shared__ unsigned short lds[2][128 * 40];

    const int tid  = threadIdx.x;
    const int lane = tid & 31;
    const int wave = tid >> 5;
    const int waveM = wave >> 1;         // 0..3
    const int waveN = wave & 1;          // 0..1
    const int blockM = blockIdx.x * 128;
    const int blockN = blockIdx.y * 128;
    const int halfSel = lane >> 4;
    const int mlane   = lane & 15;

    v8f acc[2][4] = {};

    // cooperative A loader: 2 threads / row, 32 bytes each
    const int ldrow  = tid >> 1;
    const int ldhalf = tid & 1;
    const unsigned short* gbase = A + (size_t)(blockM + ldrow) * lda + ldhalf * 16;
    unsigned short* sbase0 = &lds[0][ldrow * 40 + ldhalf * 16];
    unsigned short* sbase1 = &lds[1][ldrow * 40 + ldhalf * 16];

    // ---- prologue: stage tile kt=0 into buffer 0 ----
#if USE_ASYNC_LDS
    __builtin_amdgcn_global_load_async_to_lds_b128((gv4i_p)gbase, (sv4i_p)sbase0, 0, 0);
    __builtin_amdgcn_global_load_async_to_lds_b128((gv4i_p)gbase, (sv4i_p)sbase0, 16, 0);
    wait_async0();
#else
    {
        const uint4* src = reinterpret_cast<const uint4*>(gbase);
        uint4* dst = reinterpret_cast<uint4*>(sbase0);
        dst[0] = src[0];
        dst[1] = src[1];
    }
#endif
    __syncthreads();

    for (int kt = 0; kt < kTiles; ++kt) {
        const int cur = kt & 1;
        unsigned short* snext = cur ? sbase0 : sbase1;
        const unsigned short* lbuf = &lds[cur][0];
        const bool haveNext = (kt + 1) < kTiles;

        // ---- stage next A tile into the other buffer (overlaps with WMMAs below) ----
#if USE_ASYNC_LDS
        if (haveNext) {
            const unsigned short* gnext = gbase + (size_t)(kt + 1) * 32;
            __builtin_amdgcn_global_load_async_to_lds_b128((gv4i_p)gnext, (sv4i_p)snext, 0, 0);
            __builtin_amdgcn_global_load_async_to_lds_b128((gv4i_p)gnext, (sv4i_p)snext, 16, 0);
        }
#else
        uint4 nr0, nr1;
        if (haveNext) {
            const uint4* src = reinterpret_cast<const uint4*>(gbase + (size_t)(kt + 1) * 32);
            nr0 = src[0];
            nr1 = src[1];
        }
#endif

        // ---- B fragments (pre-packed, 2 x b128 per fragment) + prefetch of next K-step ----
        Frag b[4];
#pragma unroll
        for (int f = 0; f < 4; ++f) {
            int nTile = blockIdx.y * 8 + waveN * 4 + f;   // 16 N-tiles total
            const uint4* wsrc = reinterpret_cast<const uint4*>(
                Wpk + (((size_t)kt * 16 + nTile) * 32 + lane) * 16);
            b[f].q[0] = wsrc[0];
            b[f].q[1] = wsrc[1];
        }
        if (haveNext) {
            int nTile0 = blockIdx.y * 8 + waveN * 4;
            __builtin_prefetch(Wpk + (((size_t)(kt + 1) * 16 + nTile0) * 32 + lane) * 16, 0, 1);
        }

        // ---- A fragments from LDS + 8 WMMAs ----
#pragma unroll
        for (int r = 0; r < 2; ++r) {
            Frag a;
            int m = waveM * 32 + r * 16 + mlane;
#pragma unroll
            for (int v = 0; v < 8; ++v) {
                // A 16x32 bf16 contiguous K layout (matches packed B): K pair base
                int k0 = (halfSel * 16) + v * 2;
                a.d[v] = *reinterpret_cast<const unsigned*>(&lbuf[m * 40 + k0]);
            }
#pragma unroll
            for (int f = 0; f < 4; ++f) {
                acc[r][f] = __builtin_amdgcn_wmma_f32_16x16x32_bf16(
                    false, a.v, false, b[f].v, (short)0, acc[r][f], false, false);
            }
        }

        // ---- commit next tile, single barrier per iteration ----
#if USE_ASYNC_LDS
        if (haveNext) wait_async0();
#else
        if (haveNext) {
            uint4* dst = reinterpret_cast<uint4*>(snext);
            dst[0] = nr0;
            dst[1] = nr1;
        }
#endif
        __syncthreads();
    }

    // epilogue: C frag layout — lane: N = mlane, VGPR v: M = v + halfSel*8
#pragma unroll
    for (int r = 0; r < 2; ++r) {
#pragma unroll
        for (int f = 0; f < 4; ++f) {
            int gN = blockN + waveN * 64 + f * 16 + mlane;
            float bv = bias[gN];
#pragma unroll
            for (int v = 0; v < 8; ++v) {
                int gM = blockM + waveM * 32 + r * 16 + v + halfSel * 8;
                float val = acc[r][f][v] + bv;
                if (mode == 0) {
                    val = fmaxf(val, 0.0f);
                    outBf[(size_t)gM * 256 + gN] = f2bf(val);
                } else {
                    outF[(size_t)gM * outStride + gN] = val;
                }
            }
        }
    }
}

// ---------------- host ----------------
static inline size_t align256(size_t x) { return (x + 255) & ~(size_t)255; }

extern "C" void kernel_launch(void* const* d_in, const int* in_sizes, int n_in,
                              void* d_out, int out_size, void* d_ws, size_t ws_size,
                              hipStream_t stream) {
    const float* pts       = (const float*)d_in[0];
    const float* keypoints = (const float*)d_in[1];
    const float* trans_kp  = (const float*)d_in[2];
    const int*   neighbors = (const int*)  d_in[3];
    const float* rest_pose = (const float*)d_in[4];
    const float* latent    = (const float*)d_in[5];
    const float* W1        = (const float*)d_in[6];
    const float* b1        = (const float*)d_in[7];
    const float* W2        = (const float*)d_in[8];
    const float* b2        = (const float*)d_in[9];
    const float* W3        = (const float*)d_in[10];
    const float* b3        = (const float*)d_in[11];
    float* out = (float*)d_out;

    char* ws = (char*)d_ws;
    size_t off = 0;
    float* Rws = (float*)(ws + off);            off = align256(off + (size_t)V_KP * 9 * 4);
    unsigned short* Hin  = (unsigned short*)(ws + off); off = align256(off + (size_t)N_PTS * D_PAD * 2);
    unsigned short* H1   = (unsigned short*)(ws + off); off = align256(off + (size_t)N_PTS * D_HID * 2);
    unsigned short* H2   = (unsigned short*)(ws + off); off = align256(off + (size_t)N_PTS * D_HID * 2);
    unsigned short* Wpk1 = (unsigned short*)(ws + off); off = align256(off + (size_t)KT1  * 16 * 32 * 16 * 2);
    unsigned short* Wpk2 = (unsigned short*)(ws + off); off = align256(off + (size_t)KT23 * 16 * 32 * 16 * 2);
    unsigned short* Wpk3 = (unsigned short*)(ws + off); off = align256(off + (size_t)KT23 * 16 * 32 * 16 * 2);

    invert_kernel<<<(V_KP + 255) / 256, 256, 0, stream>>>(trans_kp, Rws, V_KP);

    repack_kernel<<<(KT1  * 512 + 255) / 256, 256, 0, stream>>>(W1, D_IN,  KT1,  Wpk1);
    repack_kernel<<<(KT23 * 512 + 255) / 256, 256, 0, stream>>>(W2, D_HID, KT23, Wpk2);
    repack_kernel<<<(KT23 * 512 + 255) / 256, 256, 0, stream>>>(W3, D_HID, KT23, Wpk3);

    feature_kernel<<<N_PTS / 256, 256, 0, stream>>>(pts, keypoints, neighbors, rest_pose,
                                                    latent, Rws, Hin, out);

    dim3 ggrid(N_PTS / 128, 2);
    gemm_wmma<<<ggrid, 256, 0, stream>>>(Hin, D_PAD, KT1,  Wpk1, b1, H1, nullptr, 0, 0);
    gemm_wmma<<<ggrid, 256, 0, stream>>>(H1,  D_HID, KT23, Wpk2, b2, H2, nullptr, 0, 0);
    gemm_wmma<<<ggrid, 256, 0, stream>>>(H2,  D_HID, KT23, Wpk3, b3, nullptr, out, D_OUT_ROW, 1);
}
